// ContrastiveLoss_2259152797845
// MI455X (gfx1250) — matrix operand order repaired and checked by reading, not compiled, not run on hardware
//
#include <hip/hip_runtime.h>
#include <hip/hip_bf16.h>
#include <math.h>

// ---------------------------------------------------------------------------
// NT-Xent (SimCLR) contrastive loss, fused, WMMA f16 path for gfx1250.
//   z = concat(z1,z2)  : 8192 x 1024 f32
//   zn = z / max(||z||, 1e-8)   (stored f16 in workspace)
//   sim = zn @ zn^T / 0.66, diag = -inf
//   loss = mean_i ( logsumexp_j sim[i,j] - sim[i, i^4096] )
// Scores are bounded by 1/T, so logsumexp uses the fixed shift SMAX = 1/T:
// no online max tracking, one exp per score, fully branchless epilogue.
// Column dimension is split across JSPLIT workgroups (partial sums per row
// accumulated in workspace) to fill all WGPs of the MI455X.
// ---------------------------------------------------------------------------

typedef __attribute__((ext_vector_type(16))) _Float16 v16h;
typedef __attribute__((ext_vector_type(8)))  _Float16 v8h;
typedef __attribute__((ext_vector_type(8)))  float    v8f;

#define N_HALF   4096
#define D        1024
#define TWO_N    8192
#define ROWS_PER_WG  64
#define WAVES_PER_WG 4
#define THREADS  (WAVES_PER_WG * 32)
#define NT       4                       // column tiles per wave per K-sweep
#define JSPLIT   4                       // column-slab split across WGs
#define COLS_PER_WG  (TWO_N / JSPLIT)    // 2048
#define JG_PER_WG    (COLS_PER_WG / (16 * NT))   // 32 j-groups per WG
#define INV_T    (1.0f / 0.66f)
#define SMAX     INV_T                   // upper bound of any score

// Zero the scalar output and the per-row partial accumulators.
__global__ __launch_bounds__(256)
void ntxent_zero(float* out, float* rowL, float* rowPos)
{
    const int i = blockIdx.x * 256 + threadIdx.x;   // 0 .. 2*TWO_N-1
    if (i < TWO_N)      rowL[i] = 0.0f;
    else                rowPos[i - TWO_N] = 0.0f;
    if (i == 0)         out[0] = 0.0f;
}

// One block per row: sum of squares -> scale -> write f16 normalized row.
__global__ __launch_bounds__(256)
void ntxent_normalize(const float* __restrict__ z1,
                      const float* __restrict__ z2,
                      _Float16* __restrict__ zn)
{
    __shared__ float red[256];
    const int row = blockIdx.x;
    const int t   = threadIdx.x;
    const float* src = (row < N_HALF) ? (z1 + (size_t)row * D)
                                      : (z2 + (size_t)(row - N_HALF) * D);
    float4 v = ((const float4*)src)[t];          // 4 floats/thread, coalesced
    red[t] = v.x * v.x + v.y * v.y + v.z * v.z + v.w * v.w;
    __syncthreads();
    for (int off = 128; off > 0; off >>= 1) {
        if (t < off) red[t] += red[t + off];
        __syncthreads();
    }
    const float scale = 1.0f / fmaxf(sqrtf(red[0]), 1e-8f);
    _Float16* dst = zn + (size_t)row * D + t * 4;
    dst[0] = (_Float16)(v.x * scale);
    dst[1] = (_Float16)(v.y * scale);
    dst[2] = (_Float16)(v.z * scale);
    dst[3] = (_Float16)(v.w * scale);
}

// Main fused GEMM + fixed-shift sum-exp kernel.
// Grid: (TWO_N/ROWS_PER_WG) x JSPLIT blocks of THREADS threads (4 wave32).
// Dynamic LDS: ROWS_PER_WG * D * 2 = 128 KB -> 2 WGs per 320 KB WGP.
__global__ __launch_bounds__(THREADS)
void ntxent_main(const _Float16* __restrict__ zn,
                 float* __restrict__ rowL, float* __restrict__ rowPos)
{
    extern __shared__ char smem[];
    _Float16* As = (_Float16*)smem;      // staged A rows: ROWS_PER_WG x D f16

    const int tid    = threadIdx.x;
    const int lane   = tid & 31;
    const int wave   = tid >> 5;
    const int hi     = (lane >> 4) & 1;  // lane half (K-split for A/B frags)
    const int n16    = lane & 15;        // tile column (N) / A-row (M) index
    const int wgRow0 = blockIdx.x * ROWS_PER_WG;
    const int jg0    = blockIdx.y * JG_PER_WG;     // this WG's column slab

    // ---- stage this WG's 64 rows of Zn into LDS (16B vector copies) ----
    {
        const uint4* gsrc = (const uint4*)(zn + (size_t)wgRow0 * D);
        uint4*       ldst = (uint4*)As;
        const int nchunks = ROWS_PER_WG * D / 8;   // 8 halfs per 16B chunk
        for (int i = tid; i < nchunks; i += THREADS) ldst[i] = gsrc[i];
    }
    __syncthreads();

    const int rowBase = wgRow0 + wave * 16;        // this wave's 16-row strip
    // A fragment source: lane holds row M = n16; hi selects K sub-halves.
    const _Float16* aRow = As + (size_t)(wave * 16 + n16) * D;

    // Per-lane state for 8 rows (cols restricted to col%16 == n16 stream).
    float l[8], pos[8];
    int   rowM[8];
#pragma unroll
    for (int i = 0; i < 8; ++i) {
        l[i]    = 0.0f;
        pos[i]  = -__builtin_inff();               // only selected / maxed
        rowM[i] = rowBase + i + hi * 8;            // C layout: M = vgpr + hi*8
    }

    for (int jg = jg0; jg < jg0 + JG_PER_WG; ++jg) {
        const int jBase = jg * 16 * NT;
        // Single per-lane B base; the NT sub-tiles live at compile-time
        // constant offsets t*16*D halfs (t*32 KB) -> immediate IOFFSETs.
        // Lane holds column N = n16 of each sub-tile, i.e. row
        // (jBase + t*16 + n16) of Zn; halves h are K = hi*16 + h (contiguous).
        const _Float16* bBase = zn + (size_t)(jBase + n16) * D + hi * 16;

        // Prefetch next j-group's first B row (global_prefetch_b8).
        if (jg + 1 < jg0 + JG_PER_WG)
            __builtin_prefetch(bBase + (size_t)16 * NT * D, 0, 0);

        v8f c[NT];
#pragma unroll
        for (int t = 0; t < NT; ++t) c[t] = (v8f){};

#pragma unroll 2
        for (int k0 = 0; k0 < D; k0 += 32) {
            union { v16h v; v8h h[2]; } a;
            // A 16x32 contiguous K layout (matches B): lo lanes K=0..15,
            // hi lanes K=16..31 (two 16B LDS loads, reused NT times).
            a.h[0] = *(const v8h*)(aRow + k0 + hi * 16);
            a.h[1] = *(const v8h*)(aRow + k0 + hi * 16 + 8);
#pragma unroll
            for (int t = 0; t < NT; ++t) {
                v16h b = *(const v16h*)(bBase + k0 + t * (16 * D));
                c[t] = __builtin_amdgcn_wmma_f32_16x16x32_f16(
                           /*neg_a=*/false, a.v, /*neg_b=*/false, b,
                           /*c_mod=*/(short)0, c[t],
                           /*reuse_a=*/false, /*reuse_b=*/false);
            }
        }

        // Branchless epilogue: scale, mask diagonal, grab positive, sum exp.
#pragma unroll
        for (int t = 0; t < NT; ++t) {
            const int col = jBase + t * 16 + n16;
#pragma unroll
            for (int i = 0; i < 8; ++i) {
                const float s = c[t][i] * INV_T;   // score in [-SMAX, SMAX]
                const int   r = rowM[i];
                float e = __expf(s - SMAX);        // arg in [-2*SMAX, 0]
                e = (col != r) ? e : 0.0f;         // diagonal mask (cndmask)
                l[i] += e;
                pos[i] = (col == (r ^ 4096)) ? s : pos[i];  // positive pair
            }
        }
    }

    // ---- combine the 16 per-lane partial (l,pos) streams per row ----
    __syncthreads();                    // done with As; reuse LDS as scratch
    float* Rl = (float*)smem;                       // [ROWS_PER_WG][16]
    float* Rp = Rl + ROWS_PER_WG * 16;
#pragma unroll
    for (int i = 0; i < 8; ++i) {
        const int rl = wave * 16 + i + hi * 8;      // local row
        Rl[rl * 16 + n16] = l[i];
        Rp[rl * 16 + n16] = pos[i];
    }
    __syncthreads();

    if (tid < ROWS_PER_WG) {
        float lt = 0.0f, pt = -__builtin_inff();
#pragma unroll
        for (int n = 0; n < 16; ++n) lt += Rl[tid * 16 + n];
#pragma unroll
        for (int n = 0; n < 16; ++n) pt = fmaxf(pt, Rp[tid * 16 + n]);
        const int r = wgRow0 + tid;
        atomicAdd(rowL + r, lt);
        // Exactly one column slab contains row r's positive; others add 0.
        atomicAdd(rowPos + r, (pt > -1e37f) ? pt : 0.0f);
    }
}

// Per-row logsumexp assembly + mean reduction to the scalar loss.
__global__ __launch_bounds__(256)
void ntxent_finalize(const float* __restrict__ rowL,
                     const float* __restrict__ rowPos,
                     float* __restrict__ out)
{
    __shared__ float red[256];
    const int t = threadIdx.x;
    const int r = blockIdx.x * 256 + t;
    const float lse = SMAX + __logf(rowL[r]);
    red[t] = (lse - rowPos[r]) * (1.0f / (float)TWO_N);
    __syncthreads();
    for (int off = 128; off > 0; off >>= 1) {
        if (t < off) red[t] += red[t + off];
        __syncthreads();
    }
    if (t == 0) atomicAdd(out, red[0]);
}

extern "C" void kernel_launch(void* const* d_in, const int* in_sizes, int n_in,
                              void* d_out, int out_size, void* d_ws, size_t ws_size,
                              hipStream_t stream)
{
    (void)in_sizes; (void)n_in; (void)out_size; (void)ws_size;
    const float* z1 = (const float*)d_in[0];
    const float* z2 = (const float*)d_in[1];
    float*       out = (float*)d_out;

    // Workspace layout: zn f16 [TWO_N*D] | rowL f32 [TWO_N] | rowPos f32 [TWO_N]
    _Float16* zn     = (_Float16*)d_ws;                    // 16 MB
    float*    rowL   = (float*)((char*)d_ws + (size_t)TWO_N * D * sizeof(_Float16));
    float*    rowPos = rowL + TWO_N;

    ntxent_zero<<<(2 * TWO_N) / 256, 256, 0, stream>>>(out, rowL, rowPos);
    ntxent_normalize<<<TWO_N, 256, 0, stream>>>(z1, z2, zn);

    const size_t ldsBytes = (size_t)ROWS_PER_WG * D * sizeof(_Float16); // 128 KB
    dim3 grid(TWO_N / ROWS_PER_WG, JSPLIT);   // 128 x 4 = 512 WGs
    ntxent_main<<<grid, THREADS, ldsBytes, stream>>>(zn, rowL, rowPos);

    ntxent_finalize<<<TWO_N / 256, 256, 0, stream>>>(rowL, rowPos, out);
}
